// VanillaLSTM_15668040695950
// MI455X (gfx1250) — compile-verified, matrix operand order checked by reading, not run on hardware
//
#include <hip/hip_runtime.h>
#include <hip/hip_bf16.h>

typedef __attribute__((ext_vector_type(16))) _Float16     v16h;
typedef __attribute__((ext_vector_type(8)))  float        v8f;
typedef __attribute__((ext_vector_type(4)))  unsigned int u32x4;

#define T_STEPS 128
#define B_TOT   8192
#define E_DIM   128
#define C_DIM   128
#define NGATES  512   // 4*C
#define KDIM    256   // E + C
#define BM      32    // batch rows per block
#define BN_EPS  1e-5f

// workspace layout (bytes)
#define WS_STATS_OFF   0        // T*8 floats: mu_u, mu_v, var_u, var_v, cov_uv
#define WS_OSTATS_OFF  8192     // T*5*2 floats: mean, invstd per (t, o)
#define WS_BIAS_OFF    16384    // 512 floats: b_ih + b_hh
#define WS_WCAT_OFF    32768    // 512*256 f16: [W_ih | W_hh] row-major (N x K)

union AV  { u32x4 q[2]; v16h h; };
union HV8 { u32x4 q; _Float16 h[8]; };

__device__ __forceinline__ float sigmoidf(float x) {
    return 1.0f / (1.0f + __expf(-x));
}

// ---------------------------------------------------------------------------
// Kernel 1: per-timestep batch statistics of the 2 raw input features.
// BN of an affine map of (u,v) needs only mean/var/cov of (u,v) over B.
// ---------------------------------------------------------------------------
__global__ void in_stats_kernel(const float* __restrict__ in,
                                float* __restrict__ stats) {
    const int t = blockIdx.x;
    const int tid = threadIdx.x;
    float su = 0.f, sv = 0.f, suu = 0.f, svv = 0.f, suv = 0.f;
    for (int b = tid; b < B_TOT; b += 256) {
        const float* p = in + (size_t)b * (T_STEPS * 2) + t * 2;
        float u = p[0], v = p[1];
        su += u; sv += v; suu += u * u; svv += v * v; suv += u * v;
    }
    __shared__ float red[256][5];
    red[tid][0] = su; red[tid][1] = sv; red[tid][2] = suu;
    red[tid][3] = svv; red[tid][4] = suv;
    __syncthreads();
    for (int s = 128; s > 0; s >>= 1) {
        if (tid < s)
            for (int j = 0; j < 5; ++j) red[tid][j] += red[tid + s][j];
        __syncthreads();
    }
    if (tid == 0) {
        const float inv = 1.0f / (float)B_TOT;
        float mu = red[0][0] * inv, mv = red[0][1] * inv;
        float* o = stats + t * 8;
        o[0] = mu;
        o[1] = mv;
        o[2] = red[0][2] * inv - mu * mu;   // var_u (biased)
        o[3] = red[0][3] * inv - mv * mv;   // var_v
        o[4] = red[0][4] * inv - mu * mv;   // cov_uv
    }
}

// ---------------------------------------------------------------------------
// Kernel 2: pack [W_ih | W_hh] into fp16 (N=512 rows, K=256 cols, row-major)
// and fuse b_ih + b_hh.
// ---------------------------------------------------------------------------
__global__ void prep_w_kernel(const float* __restrict__ W_ih,
                              const float* __restrict__ W_hh,
                              const float* __restrict__ b_ih,
                              const float* __restrict__ b_hh,
                              _Float16* __restrict__ Wcat,
                              float* __restrict__ bias) {
    const int n = blockIdx.x;    // 0..511
    const int k = threadIdx.x;   // 0..255
    float w = (k < C_DIM) ? W_ih[n * C_DIM + k] : W_hh[n * C_DIM + (k - C_DIM)];
    Wcat[n * KDIM + k] = (_Float16)w;
    if (k == 0) bias[n] = b_ih[n] + b_hh[n];
}

// ---------------------------------------------------------------------------
// Kernel 3: fused embedding + per-step BN/ReLU + LSTM recurrence (WMMA) +
// output projection (pre-BN values into d_out).
// Block: 256 threads = 8 waves = 2 row-strips x 4 channel-groups.
// Each block owns BM=32 batch rows for all 128 time steps.
// ---------------------------------------------------------------------------
__global__ void __launch_bounds__(256)
lstm_kernel(const float* __restrict__ in,
            const float* __restrict__ W_emb,
            const float* __restrict__ g_emb,
            const float* __restrict__ be_emb,
            const float* __restrict__ stats,
            const _Float16* __restrict__ Wcat,
            const float* __restrict__ bias,
            const float* __restrict__ W_out,
            float* __restrict__ out) {
    __shared__ __align__(32) _Float16 xh[BM][KDIM];   // [x_t | h] fp16 tile
    __shared__ float A0[E_DIM], A1[E_DIM], Be[E_DIM]; // per-t embed BN coeffs
    __shared__ float du_s[BM], dv_s[BM];
    __shared__ float Wout_s[5][C_DIM];

    const int tid    = threadIdx.x;
    const int lane   = tid & 31;
    const int wid    = tid >> 5;
    const int laneLo = lane & 15;
    const int laneHi = lane >> 4;          // 0: lanes 0-15, 1: lanes 16-31
    const int r      = wid >> 2;           // row strip (0..1) -> rows 16r..16r+15
    const int g      = wid & 3;            // channel group (0..3) -> 32 channels
    const int b0     = blockIdx.x * BM;

    // ---- one-time init ----
    for (int i = tid; i < 5 * C_DIM; i += 256)
        Wout_s[i / C_DIM][i % C_DIM] = W_out[i];
    for (int i = tid; i < E_DIM; i += 256) Be[i] = be_emb[i];
    for (int i = tid; i < BM * C_DIM; i += 256)           // h_0 = 0
        xh[i >> 7][C_DIM + (i & 127)] = (_Float16)0.f;

    // gate bias per lane: tile nt = 8*gate + 2*g + p, column = laneLo
    float biasreg[8];
#pragma unroll
    for (int j = 0; j < 8; ++j) {
        int nt = 8 * (j >> 1) + 2 * g + (j & 1);
        biasreg[j] = bias[nt * 16 + laneLo];
    }

    float creg[2][8];                                     // c_0 = 0 (fp32)
#pragma unroll
    for (int p = 0; p < 2; ++p)
#pragma unroll
        for (int v = 0; v < 8; ++v) creg[p][v] = 0.f;

    for (int t = 0; t < T_STEPS; ++t) {
        // ---- stage per-t embedding BN coefficients & per-row deltas ----
        if (tid < E_DIM) {
            const float* st = stats + t * 8;
            float w0 = W_emb[tid * 2 + 0], w1 = W_emb[tid * 2 + 1];
            float var = w0 * w0 * st[2] + w1 * w1 * st[3] + 2.f * w0 * w1 * st[4];
            float is = rsqrtf(var + BN_EPS) * g_emb[tid];
            A0[tid] = w0 * is;
            A1[tid] = w1 * is;
        }
        if (tid < BM) {
            const float* st = stats + t * 8;
            const float* p = in + (size_t)(b0 + tid) * (T_STEPS * 2) + t * 2;
            du_s[tid] = p[0] - st[0];
            dv_s[tid] = p[1] - st[1];
        }
        __syncthreads();

        // ---- build x_t tile (embed + BN + ReLU) into xh[:, 0:128] fp16 ----
        for (int i = tid; i < BM * E_DIM; i += 256) {
            int row = i >> 7, e = i & 127;
            float val = du_s[row] * A0[e] + dv_s[row] * A1[e] + Be[e];
            xh[row][e] = (_Float16)fmaxf(val, 0.f);
        }
        __syncthreads();

        // ---- GEMM: gates[32,512] = xh[32,256] x Wcat^T via WMMA ----
        // Launder a ZERO OFFSET (not the pointer) each iteration:
        //  * loop-variant offset => LICM cannot hoist the 64 B-tiles out of
        //    the t-loop (which previously spilled 512 VGPRs to scratch);
        //  * pointer provenance is preserved => backend keeps the global
        //    address space and emits global_load_b128 (not flat_load_b128,
        //    which would tie LOADcnt to DScnt and serialize against LDS).
        size_t woff = 0;
        asm volatile("" : "+s"(woff));
        const _Float16* __restrict__ Wv = Wcat + woff;

        v8f acc[8];
#pragma unroll
        for (int j = 0; j < 8; ++j) acc[j] = (v8f){0,0,0,0,0,0,0,0};

        for (int kk = 0; kk < 8; ++kk) {
            // A tile 16x32 f16: lanes 0-15 K 0..7/16..23, lanes 16-31 K 8..15/24..31
            const _Float16* pa = &xh[r * 16 + laneLo][kk * 32 + laneHi * 8];
            AV a;
            a.q[0] = *(const u32x4*)(pa);
            a.q[1] = *(const u32x4*)(pa + 16);
#pragma unroll
            for (int j = 0; j < 8; ++j) {
                int nt = 8 * (j >> 1) + 2 * g + (j & 1);
                // 32-bit element offset -> saddr + voffset addressing
                int eoff = (nt * 16 + laneLo) * KDIM + kk * 32 + laneHi * 8;
                const _Float16* pb = Wv + eoff;
                AV bv;
                bv.q[0] = *(const u32x4*)(pb);
                bv.q[1] = *(const u32x4*)(pb + 16);
                acc[j] = __builtin_amdgcn_wmma_f32_16x16x32_f16(
                    false, a.h, false, bv.h, (short)0, acc[j], false, false);
            }
        }

        // ---- LSTM cell update, entirely in registers ----
        float hreg[2][8];
#pragma unroll
        for (int p = 0; p < 2; ++p) {
#pragma unroll
            for (int v = 0; v < 8; ++v) {
                float ig = sigmoidf(acc[0 + p][v] + biasreg[0 + p]);
                float fg = sigmoidf(acc[2 + p][v] + biasreg[2 + p]);
                float gg = tanhf   (acc[4 + p][v] + biasreg[4 + p]);
                float og = sigmoidf(acc[6 + p][v] + biasreg[6 + p]);
                float c  = fg * creg[p][v] + ig * gg;
                creg[p][v] = c;
                hreg[p][v] = og * tanhf(c);
            }
        }
        __syncthreads();   // all reads of xh for step t complete

        // ---- publish h_t (fp16) into xh[:, 128:256] ----
#pragma unroll
        for (int p = 0; p < 2; ++p)
#pragma unroll
            for (int v = 0; v < 8; ++v)
                xh[r * 16 + v + 8 * laneHi][C_DIM + g * 32 + p * 16 + laneLo] =
                    (_Float16)hreg[p][v];
        __syncthreads();   // h_t visible to all waves

        // ---- output projection: y_pre[b, t, o] = h_t . W_out[o]  (N=5) ----
        {
            int row = tid & 31;
            int o   = tid >> 5;
            if (o < 5) {
                const u32x4* hp = (const u32x4*)&xh[row][C_DIM];
                float s = 0.f;
                for (int cc = 0; cc < C_DIM / 8; ++cc) {   // ds_load_b128 each
                    HV8 hv; hv.q = hp[cc];
#pragma unroll
                    for (int e = 0; e < 8; ++e)
                        s += (float)hv.h[e] * Wout_s[o][cc * 8 + e];
                }
                out[((size_t)(b0 + row) * T_STEPS + t) * 5 + o] = s;
            }
        }
        // next iteration's stage/build barriers fence the xh x-part overwrite
    }
}

// ---------------------------------------------------------------------------
// Kernel 4: per-(t, o) mean / inv-std over the batch (output BN stats).
// ---------------------------------------------------------------------------
__global__ void out_stats_kernel(const float* __restrict__ y,
                                 float* __restrict__ ostats) {
    const int t = blockIdx.x;
    const int tid = threadIdx.x;
    float s[5] = {0, 0, 0, 0, 0}, sq[5] = {0, 0, 0, 0, 0};
    for (int b = tid; b < B_TOT; b += 256) {
        const float* p = y + ((size_t)b * T_STEPS + t) * 5;
        for (int o = 0; o < 5; ++o) {
            float v = p[o];
            s[o] += v; sq[o] += v * v;
        }
    }
    __shared__ float red[256][10];
    for (int o = 0; o < 5; ++o) { red[tid][o] = s[o]; red[tid][5 + o] = sq[o]; }
    __syncthreads();
    for (int st = 128; st > 0; st >>= 1) {
        if (tid < st)
            for (int j = 0; j < 10; ++j) red[tid][j] += red[tid + st][j];
        __syncthreads();
    }
    if (tid == 0) {
        const float inv = 1.0f / (float)B_TOT;
        for (int o = 0; o < 5; ++o) {
            float m = red[0][o] * inv;
            float var = red[0][5 + o] * inv - m * m;
            ostats[t * 10 + o * 2 + 0] = m;
            ostats[t * 10 + o * 2 + 1] = rsqrtf(var + BN_EPS);
        }
    }
}

// ---------------------------------------------------------------------------
// Kernel 5: in-place normalize + affine + ReLU of d_out.
// ---------------------------------------------------------------------------
__global__ void out_norm_kernel(float* __restrict__ y,
                                const float* __restrict__ ostats,
                                const float* __restrict__ g_out,
                                const float* __restrict__ be_out) {
    size_t idx = (size_t)blockIdx.x * 256 + threadIdx.x;
    const size_t N = (size_t)B_TOT * T_STEPS * 5;
    if (idx >= N) return;
    int rem = (int)(idx % (T_STEPS * 5));
    int t = rem / 5, o = rem % 5;
    float v = y[idx];
    v = (v - ostats[t * 10 + o * 2]) * ostats[t * 10 + o * 2 + 1] * g_out[o] +
        be_out[o];
    y[idx] = fmaxf(v, 0.f);
}

extern "C" void kernel_launch(void* const* d_in, const int* in_sizes, int n_in,
                              void* d_out, int out_size, void* d_ws, size_t ws_size,
                              hipStream_t stream) {
    const float* in     = (const float*)d_in[0];
    const float* W_emb  = (const float*)d_in[1];
    // d_in[2] = b_emb : cancelled by the batch-norm that follows it
    const float* g_emb  = (const float*)d_in[3];
    const float* be_emb = (const float*)d_in[4];
    const float* W_ih   = (const float*)d_in[5];
    const float* W_hh   = (const float*)d_in[6];
    const float* b_ih   = (const float*)d_in[7];
    const float* b_hh   = (const float*)d_in[8];
    const float* W_out  = (const float*)d_in[9];
    // d_in[10] = b_out : cancelled by the batch-norm that follows it
    const float* g_out  = (const float*)d_in[11];
    const float* be_out = (const float*)d_in[12];

    float* out = (float*)d_out;
    char*  ws  = (char*)d_ws;
    float*     stats  = (float*)(ws + WS_STATS_OFF);
    float*     ostats = (float*)(ws + WS_OSTATS_OFF);
    float*     bias   = (float*)(ws + WS_BIAS_OFF);
    _Float16*  Wcat   = (_Float16*)(ws + WS_WCAT_OFF);

    in_stats_kernel<<<T_STEPS, 256, 0, stream>>>(in, stats);
    prep_w_kernel<<<NGATES, 256, 0, stream>>>(W_ih, W_hh, b_ih, b_hh, Wcat, bias);
    lstm_kernel<<<B_TOT / BM, 256, 0, stream>>>(in, W_emb, g_emb, be_emb, stats,
                                                Wcat, bias, W_out, out);
    out_stats_kernel<<<T_STEPS, 256, 0, stream>>>(out, ostats);
    const int N = B_TOT * T_STEPS * 5;
    out_norm_kernel<<<(N + 255) / 256, 256, 0, stream>>>(out, ostats, g_out, be_out);
}
